// HandMadeLSTM_54331336294795
// MI455X (gfx1250) — matrix operand order repaired and checked by reading, not compiled, not run on hardware
//
#include <hip/hip_runtime.h>
#include <hip/hip_bf16.h>

// ---------------------------------------------------------------------------
// Bidirectional LSTM, B=64 T=512 I=H=512, fp32 in/out, bf16 WMMA compute.
// Persistent fused kernel: 2 directions x 16 WGPs, each WGP owns 32 hidden
// units. Per step: gates = [x_t | h] @ [W_ih|W_hh]^T + b  via
// v_wmma_f32_16x16x32_bf16. x_t tile staged into LDS by the Tensor Data
// Mover (tensor_load_to_lds + s_wait_tensorcnt), the cross-WGP h exchange
// uses global_load_async_to_lds_b128 (+ s_wait_asynccnt) through an
// L2-resident parity double buffer, one agent-scope barrier per timestep.
// ---------------------------------------------------------------------------

typedef __attribute__((ext_vector_type(16))) __bf16 v16bf;
typedef __attribute__((ext_vector_type(8)))  __bf16 v8bf;
typedef __attribute__((ext_vector_type(8)))  float  v8f;
typedef unsigned __attribute__((ext_vector_type(4))) v4u;
typedef int      __attribute__((ext_vector_type(4))) v4i;
typedef int      __attribute__((ext_vector_type(8))) v8i;

#define BSZ 64
#define TSZ 512
#define ISZ 512
#define HSZ 512
#define GSZ 2048   // 4*H  (gate order i,f,g,o)
#define NWGP 16    // workgroups per direction
#define JPW 32     // hidden units per workgroup (512/16)
#define NTHREADS 256  // 8 waves (wave32)
#define LPAD 8     // LDS row padding (elements)

#if __has_builtin(__builtin_amdgcn_tensor_load_to_lds) && \
    __has_builtin(__builtin_amdgcn_s_wait_tensorcnt)
#define HAVE_TDM 1
#else
#define HAVE_TDM 0
#endif

#if __has_builtin(__builtin_amdgcn_global_load_async_to_lds_b128)
#define HAVE_ASYNC 1
#else
#define HAVE_ASYNC 0
#endif

// AS-qualified pointer types for the async/TDM builtins
typedef __attribute__((address_space(1))) v4i gas_v4i;   // global
typedef __attribute__((address_space(3))) v4i las_v4i;   // LDS

struct DirBar { unsigned count; unsigned epoch; unsigned pad[62]; }; // 256B

__device__ __forceinline__ v16bf cat16(v8bf lo, v8bf hi) {
  return __builtin_shufflevector(lo, hi, 0,1,2,3,4,5,6,7,8,9,10,11,12,13,14,15);
}

#if HAVE_TDM || HAVE_ASYNC
// 32-bit LDS offset of a generic pointer to __shared__ (flat->local keeps low bits)
__device__ __forceinline__ unsigned lds_addr_of(void* p) {
  return (unsigned)(unsigned long long)(__attribute__((address_space(3))) char*)p;
}
#endif

__device__ __forceinline__ void wait_asynccnt0() {
#if __has_builtin(__builtin_amdgcn_s_wait_asynccnt)
  __builtin_amdgcn_s_wait_asynccnt(0);
#else
  asm volatile("s_wait_asynccnt 0x0" ::: "memory");
#endif
}

// ---- one-time pack: weights to bf16 [2][2048][1024] (k<512: W_ih, else W_hh),
// ---- biases folded to fp32 [2][2048], xs converted to bf16 [B][T][I].
__global__ void lstm_prep_kernel(
    const float* __restrict__ xs,
    const float* __restrict__ Wih_f, const float* __restrict__ Whh_f,
    const float* __restrict__ bih_f, const float* __restrict__ bhh_f,
    const float* __restrict__ Wih_b, const float* __restrict__ Whh_b,
    const float* __restrict__ bih_b, const float* __restrict__ bhh_b,
    __bf16* __restrict__ Wcat, float* __restrict__ bias,
    __bf16* __restrict__ xsbf)
{
  const int stride = gridDim.x * blockDim.x;
  const int tid0 = blockIdx.x * blockDim.x + threadIdx.x;

  const int totalW = 2 * GSZ * (ISZ + HSZ);
  for (int i = tid0; i < totalW; i += stride) {
    const int d = i / (GSZ * (ISZ + HSZ));
    const int r = (i / (ISZ + HSZ)) % GSZ;
    const int k = i % (ISZ + HSZ);
    const float* Wih = d ? Wih_b : Wih_f;
    const float* Whh = d ? Whh_b : Whh_f;
    const float v = (k < ISZ) ? Wih[(size_t)r * ISZ + k]
                              : Whh[(size_t)r * HSZ + (k - ISZ)];
    Wcat[i] = (__bf16)v;
  }
  for (int i = tid0; i < 2 * GSZ; i += stride) {
    const int d = i / GSZ; const int r = i % GSZ;
    bias[i] = d ? (bih_b[r] + bhh_b[r]) : (bih_f[r] + bhh_f[r]);
  }
  const int nxch = BSZ * TSZ * ISZ / 4;
  for (int i = tid0; i < nxch; i += stride) {
    const float4 v = *(const float4*)(xs + (size_t)i * 4);
    xsbf[(size_t)i * 4 + 0] = (__bf16)v.x;
    xsbf[(size_t)i * 4 + 1] = (__bf16)v.y;
    xsbf[(size_t)i * 4 + 2] = (__bf16)v.z;
    xsbf[(size_t)i * 4 + 3] = (__bf16)v.w;
  }
}

__global__ __launch_bounds__(NTHREADS, 1)
void lstm_persistent_kernel(
    const __bf16* __restrict__ xsbf,   // [B][T][I] bf16
    const __bf16* __restrict__ Wcat,   // [2][2048][1024] bf16
    const float*  __restrict__ bias,   // [2][2048] fp32
    __bf16* __restrict__ Hglob,        // [2 dir][2 parity][64][512] bf16
    DirBar* __restrict__ bars,         // [2]
    float* __restrict__ out)           // [B][T][2H] fp32
{
  const int dir  = blockIdx.x >> 4;
  const int wgp  = blockIdx.x & 15;
  const int j0   = wgp * JPW;                 // first hidden unit owned
  const int tid  = threadIdx.x;
  const int lane = tid & 31;
  const int wave = tid >> 5;
  const int gt    = wave >> 1;                // gate type 0..3
  const int nsub  = wave & 1;                 // which 16-col subtile
  const int nlane = lane & 15;
  const int khalf = lane >> 4;                // 0 or 1

  // ~293 KB static LDS (<= 320 KB/WGP)
  __shared__ __bf16 Xbuf[BSZ][ISZ + LPAD];        // x_t tile (TDM dest), 65 KB
  __shared__ __bf16 Hbuf[BSZ][HSZ + LPAD];        // h tile (async dest),  65 KB
  __shared__ __bf16 WhhL[4 * JPW][HSZ + LPAD];    // W_hh slice, 130 KB
  __shared__ float  Gbuf[BSZ][4 * JPW + 4];       // gate exchange, 33 KB

  const __bf16* WcatD = Wcat + (size_t)dir * GSZ * (ISZ + HSZ);
  const float*  biasD = bias + dir * GSZ;
  DirBar* bar = bars + dir;

  // ---- one-time: stage recurrent-weight slice (128 gate rows x K 512..1023)
  for (int idx = tid; idx < (4 * JPW) * (HSZ / 8); idx += NTHREADS) {
    const int lcol = idx >> 6;                // 64 x 16B chunks per row
    const int off  = (idx & 63) * 8;
    const int g  = lcol >> 5;
    const int jl = lcol & (JPW - 1);
    const int gcol = g * HSZ + j0 + jl;
    const __bf16* src = WcatD + (size_t)gcol * (ISZ + HSZ) + ISZ + off;
    *(uint4*)&WhhL[lcol][off] = *(const uint4*)src;
  }
  // ---- one-time: h0 = 0
  {
    const uint4 z4 = {0u, 0u, 0u, 0u};
    for (int idx = tid; idx < BSZ * (HSZ / 8); idx += NTHREADS) {
      const int b = idx >> 6; const int off = (idx & 63) * 8;
      *(uint4*)&Hbuf[b][off] = z4;
    }
  }

  // per-thread cell state: batch row bown, 8 consecutive local hidden cols
  const int bown   = tid >> 2;
  const int jlbase = (tid & 3) * 8;
  float creg[8];
  #pragma unroll
  for (int q = 0; q < 8; ++q) creg[q] = 0.f;

  const int lcolbase = gt * JPW + nsub * 16;          // local gate column base
  const int gcolbase = gt * HSZ + j0 + nsub * 16;     // global gate column base
  const float bv = biasD[gcolbase + nlane];

  for (int t = 0; t < TSZ; ++t) {
    const int tphys = dir ? (TSZ - 1 - t) : t;

    // ---- stage x_t tile [64 x 512] bf16 into Xbuf ----
#if HAVE_TDM
    if (wave == 0) {
      // Tensor DMA descriptor: 2D tile, rows = batch (global stride T*I elems),
      // LDS pad 4 dwords after every 256 dwords -> 520-element padded rows.
      const unsigned long long ga =
          (unsigned long long)(const void*)(xsbf + (size_t)tphys * ISZ);
      v4u g0; v8i g1; v4i g2 = {0, 0, 0, 0}; v4i g3 = {0, 0, 0, 0};
      g0.x = 1u;                                  // count=1 valid descriptor
      g0.y = lds_addr_of(&Xbuf[0][0]);            // lds_addr
      g0.z = (unsigned)ga;                        // global_addr[31:0]
      g0.w = (unsigned)(ga >> 32) | (2u << 30);   // global_addr[56:32] | type=2
      g1[0] = (int)((1u << 16)      // data_size = 2 bytes
                  | (1u << 20)      // pad_enable
                  | (7u << 22)      // pad_interval = 256 dwords
                  | (3u << 25));    // pad_amount = 4 dwords
      g1[1] = (int)(512u << 16);    // tensor_dim0 = 512 (lo16)
      g1[2] = (int)(64u << 16);     // tensor_dim0 hi=0 | tensor_dim1 = 64 (lo16)
      g1[3] = (int)(512u << 16);    // tensor_dim1 hi=0 | tile_dim0 = 512
      g1[4] = 64;                   // tile_dim1 = 64 | tile_dim2 = 0
      g1[5] = TSZ * ISZ;            // tensor_dim0_stride lo32 = 262144
      g1[6] = 0;                    // stride hi | tensor_dim1_stride lo
      g1[7] = 0;
#if defined(__clang_major__) && __clang_major__ >= 23
      v8i g4 = {0, 0, 0, 0, 0, 0, 0, 0};
      __builtin_amdgcn_tensor_load_to_lds(g0, g1, g2, g3, g4, 0);
#else
      __builtin_amdgcn_tensor_load_to_lds(g0, g1, g2, g3, 0);
#endif
      __builtin_amdgcn_s_wait_tensorcnt(0);
    }
#else
    for (int idx = tid; idx < BSZ * (ISZ / 8); idx += NTHREADS) {
      const int b = idx >> 6; const int off = (idx & 63) * 8;
      *(uint4*)&Xbuf[b][off] =
          *(const uint4*)&xsbf[((size_t)b * TSZ + tphys) * ISZ + off];
    }
#endif
    __syncthreads();

    // ---- gates: this wave computes 64 rows x 16 cols, K = 1024
    v8f acc[4] = {};

    // Phase 1: K = 0..511 (x part), B fragments streamed from L2-resident W_ih
    #pragma unroll 2
    for (int kt = 0; kt < 16; ++kt) {
      const int k0 = kt * 32;
      const __bf16* wp =
          WcatD + (size_t)(gcolbase + nlane) * (ISZ + HSZ) + k0 + khalf * 16;
      const v16bf bfrag = cat16(*(const v8bf*)wp, *(const v8bf*)(wp + 8));
      #pragma unroll
      for (int mt = 0; mt < 4; ++mt) {
        const int row = mt * 16 + nlane;
        const v16bf afrag = cat16(*(const v8bf*)&Xbuf[row][k0 + khalf * 16],
                                  *(const v8bf*)&Xbuf[row][k0 + khalf * 16 + 8]);
        acc[mt] = __builtin_amdgcn_wmma_f32_16x16x32_bf16(
            false, afrag, false, bfrag, (short)0, acc[mt], false, false);
      }
    }
    // Phase 2: K = 512..1023 (h part), B fragments from LDS-resident W_hh slice
    #pragma unroll 2
    for (int kt = 0; kt < 16; ++kt) {
      const int k0 = kt * 32;
      const __bf16* wp = &WhhL[lcolbase + nlane][k0 + khalf * 16];
      const v16bf bfrag = cat16(*(const v8bf*)wp, *(const v8bf*)(wp + 8));
      #pragma unroll
      for (int mt = 0; mt < 4; ++mt) {
        const int row = mt * 16 + nlane;
        const v16bf afrag = cat16(*(const v8bf*)&Hbuf[row][k0 + khalf * 16],
                                  *(const v8bf*)&Hbuf[row][k0 + khalf * 16 + 8]);
        acc[mt] = __builtin_amdgcn_wmma_f32_16x16x32_bf16(
            false, afrag, false, bfrag, (short)0, acc[mt], false, false);
      }
    }

    // ---- +bias, exchange gate tiles via LDS (C layout: M = vgpr + 8*(lane>=16))
    #pragma unroll
    for (int mt = 0; mt < 4; ++mt) {
      #pragma unroll
      for (int r = 0; r < 8; ++r) {
        const int row = mt * 16 + r + (khalf ? 8 : 0);
        Gbuf[row][lcolbase + nlane] = acc[mt][r] + bv;
      }
    }
    __syncthreads();

    // ---- elementwise cell update (8 elements per thread, c lives in VGPRs)
    __bf16* HgW = Hglob + ((size_t)(dir * 2 + (t & 1))) * BSZ * HSZ;
    #pragma unroll
    for (int q = 0; q < 8; ++q) {
      const int jl = jlbase + q;
      const float ig = Gbuf[bown][jl];
      const float fg = Gbuf[bown][JPW + jl];
      const float gg = Gbuf[bown][2 * JPW + jl];
      const float og = Gbuf[bown][3 * JPW + jl];
      const float si = 1.f / (1.f + __expf(-ig));
      const float sf = 1.f / (1.f + __expf(-fg));
      const float so = 1.f / (1.f + __expf(-og));
      const float c  = sf * creg[q] + si * tanhf(gg);
      creg[q] = c;
      const float h = so * tanhf(c);
      HgW[bown * HSZ + j0 + jl] = (__bf16)h;
      out[((size_t)bown * TSZ + tphys) * (2 * HSZ) + dir * HSZ + (j0 + jl)] = h;
    }

    // ---- publish h slice, sync the 16 WGPs of this direction
    __threadfence();
    __syncthreads();
    if (tid == 0) {
      const unsigned prev = __hip_atomic_fetch_add(
          &bar->count, 1u, __ATOMIC_ACQ_REL, __HIP_MEMORY_SCOPE_AGENT);
      if (prev == NWGP - 1) {
        __hip_atomic_store(&bar->count, 0u, __ATOMIC_RELAXED,
                           __HIP_MEMORY_SCOPE_AGENT);
        __hip_atomic_fetch_add(&bar->epoch, 1u, __ATOMIC_RELEASE,
                               __HIP_MEMORY_SCOPE_AGENT);
      } else {
        while (__hip_atomic_load(&bar->epoch, __ATOMIC_ACQUIRE,
                                 __HIP_MEMORY_SCOPE_AGENT) <= (unsigned)t)
          __builtin_amdgcn_s_sleep(1);
      }
    }
    __builtin_amdgcn_s_cluster_barrier();
    __syncthreads();

    // ---- gather full h (parity buffer of step t) into Hbuf
#if HAVE_ASYNC
    for (int idx = tid; idx < BSZ * (HSZ / 8); idx += NTHREADS) {
      const int b = idx >> 6; const int off = (idx & 63) * 8;
      __builtin_amdgcn_global_load_async_to_lds_b128(
          (gas_v4i*)&HgW[b * HSZ + off],
          (las_v4i*)&Hbuf[b][off], 0, 0);
    }
    wait_asynccnt0();
#else
    for (int idx = tid; idx < BSZ * (HSZ / 8); idx += NTHREADS) {
      const int b = idx >> 6; const int off = (idx & 63) * 8;
      *(uint4*)&Hbuf[b][off] = *(const uint4*)&HgW[b * HSZ + off];
    }
#endif
    // next iteration's __syncthreads() (after x staging) covers this copy
  }
}

extern "C" void kernel_launch(void* const* d_in, const int* in_sizes, int n_in,
                              void* d_out, int out_size, void* d_ws, size_t ws_size,
                              hipStream_t stream)
{
  (void)in_sizes; (void)n_in; (void)out_size; (void)ws_size;

  const float* xs    = (const float*)d_in[0];
  const float* Wih_f = (const float*)d_in[1];
  const float* Whh_f = (const float*)d_in[2];
  const float* bih_f = (const float*)d_in[3];
  const float* bhh_f = (const float*)d_in[4];
  const float* Wih_b = (const float*)d_in[5];
  const float* Whh_b = (const float*)d_in[6];
  const float* bih_b = (const float*)d_in[7];
  const float* bhh_b = (const float*)d_in[8];

  char* ws = (char*)d_ws;
  __bf16* Wcat  = (__bf16*)(ws);                                         // 8 MiB
  float*  bias  = (float*)(ws + (size_t)8 * 1024 * 1024);                // 16 KiB
  __bf16* Hglob = (__bf16*)(ws + (size_t)8 * 1024 * 1024 + 64 * 1024);   // 256 KiB
  DirBar* bars  = (DirBar*)(ws + (size_t)8 * 1024 * 1024 + 512 * 1024);  // 512 B
  __bf16* xsbf  = (__bf16*)(ws + (size_t)16 * 1024 * 1024);              // 32 MiB

  (void)hipMemsetAsync(bars, 0, 2 * sizeof(DirBar), stream);

  lstm_prep_kernel<<<256, 256, 0, stream>>>(
      xs, Wih_f, Whh_f, bih_f, bhh_f, Wih_b, Whh_b, bih_b, bhh_b,
      Wcat, bias, xsbf);

  lstm_persistent_kernel<<<2 * NWGP, NTHREADS, 0, stream>>>(
      xsbf, Wcat, bias, Hglob, bars, (float*)d_out);
}
